// Encoder_38397007626389
// MI455X (gfx1250) — compile-verified, matrix operand order checked by reading, not run on hardware
//
#include <hip/hip_runtime.h>
#include <stdint.h>

// ---------------- problem constants (from reference) ----------------
#define N_ROOM 20000
#define N_FURN 50000
#define NE_FF  400000
#define NE_RR  200000
#define NE_RF  400000
#define ROOM_D 14
#define LEAKY_SLOPE 0.01f

// ---------------- WMMA types (gfx1250, wave32) ----------------
typedef __bf16 bf16;
typedef bf16  v16bf  __attribute__((ext_vector_type(16)));
typedef bf16  bf16x2 __attribute__((ext_vector_type(2)));
typedef float v8f    __attribute__((ext_vector_type(8)));

__device__ __forceinline__ unsigned pack_bf16(float lo, float hi) {
  bf16x2 h = { (bf16)lo, (bf16)hi };           // v_cvt_pk_bf16_f32 path
  return __builtin_bit_cast(unsigned, h);
}

// =====================================================================
// Tiled GEMM: C[M,N] = act(A16[M,K] @ Bt16[N,K]^T + bias)
// A and B are pre-converted bf16 (B pre-transposed to [N][K]).
// 256 threads = 8 waves; WG tile 128x64; each wave 2x2 of 16x16 tiles
//   => 4 x v_wmma_f32_16x16x32_bf16 per wave per K-step.
// Interior tiles: staged with global_load_async_to_lds_b128 (ASYNCcnt),
// overlapped with WMMA on the other LDS buffer. Edge tiles / K%8!=0:
// branchless clamped bf16-pair staging.
// =====================================================================
#define TM 128
#define TN 64
#define TK 32
#define LDST 40   // bf16 row stride: 80B -> 16B-aligned rows, conflict-free

__global__ __launch_bounds__(256) void gemm_bf16_wmma(
    const bf16* __restrict__ A, const bf16* __restrict__ Bt,
    float* __restrict__ C, int M, int N, int K,
    const float* __restrict__ bias, int act /*0 none,1 relu,2 tanh*/)
{
  __shared__ __attribute__((aligned(16))) bf16 sA[2][TM][LDST]; // [m][k]
  __shared__ __attribute__((aligned(16))) bf16 sB[2][TN][LDST]; // [n][k]
  const int tid  = threadIdx.x;
  const int wave = tid >> 5;
  const int lane = tid & 31;
  const int half = lane >> 4;        // K-group select (ISA 7.12.2)
  const int l16  = lane & 15;
  const int wMi = wave & 3;          // 4 M-groups of 32 rows
  const int wNi = wave >> 2;         // 2 N-groups of 32 cols
  const int bM = blockIdx.x * TM;
  const int bN = blockIdx.y * TN;

  v8f acc[2][2] = {};

  // ---- async staging: 128x32 A tile (2 x b128/thread), 64x32 B tile (1 x) ----
  auto stageAsync = [&](int b, int k0) {
#pragma unroll
    for (int i = 0; i < 2; ++i) {
      int q = tid + (i << 8);
      int r = q >> 2, c = (q & 3) << 3;                  // c in bf16 units (16B)
      const bf16* src = A + (size_t)(bM + r) * K + k0 + c;
      unsigned dst = (unsigned)(uintptr_t)&sA[b][r][c];  // low 32b = LDS addr
      asm volatile("global_load_async_to_lds_b128 %0, %1, off"
                   :: "v"(dst), "v"(src) : "memory");
    }
    {
      int r = tid >> 2, c = (tid & 3) << 3;
      const bf16* src = Bt + (size_t)(bN + r) * K + k0 + c;
      unsigned dst = (unsigned)(uintptr_t)&sB[b][r][c];
      asm volatile("global_load_async_to_lds_b128 %0, %1, off"
                   :: "v"(dst), "v"(src) : "memory");
    }
  };

  // ---- guarded staging: clamped addresses + select-zero, bf16 pairs ----
  auto stageGuard = [&](int b, int k0) {
#pragma unroll
    for (int i = 0; i < 8; ++i) {            // A: 2048 pairs, 8/thread
      int q = tid + (i << 8);
      int r = q >> 4, c = q & 15;
      int gr = bM + r, gk = k0 + (c << 1);
      int grc = gr < M ? gr : M - 1;
      int gkc = gk < K ? gk : 0;
      unsigned v = *(const unsigned*)&A[(size_t)grc * K + gkc]; // K even
      *(unsigned*)&sA[b][r][c << 1] = (gr < M && gk < K) ? v : 0u;
    }
#pragma unroll
    for (int i = 0; i < 4; ++i) {            // B: 1024 pairs, 4/thread
      int q = tid + (i << 8);
      int nn = q & 63, r = q >> 6;
      int gn = bN + nn, gk = k0 + (r << 1);
      int gnc = gn < N ? gn : N - 1;
      int gkc = gk < K ? gk : 0;
      unsigned v = *(const unsigned*)&Bt[(size_t)gnc * K + gkc];
      *(unsigned*)&sB[b][nn][r << 1] = (gn < N && gk < K) ? v : 0u;
    }
  };

  const bool interior = (bM + TM <= M) && (bN + TN <= N) && ((K & 7) == 0);
  auto stage = [&](int b, int k0) {
    if (interior && (k0 + TK <= K)) stageAsync(b, k0);
    else                            stageGuard(b, k0);
  };

  auto compute = [&](int b) {
    union F { v16bf v; unsigned u[8]; };
    F fa[2], fb[2];
#pragma unroll
    for (int t = 0; t < 2; ++t) {
      int rowA = (wMi << 5) + (t << 4) + l16;
      int rowB = (wNi << 5) + (t << 4) + l16;
#pragma unroll
      for (int i = 0; i < 8; ++i) {
        // VGPR i holds K-pair base = 16*(i/4) + 8*half + 2*(i%4)
        int kk = ((i >> 2) << 4) + (half << 3) + ((i & 3) << 1);
        fa[t].u[i] = *(const unsigned*)&sA[b][rowA][kk];
        fb[t].u[i] = *(const unsigned*)&sB[b][rowB][kk];
      }
    }
#pragma unroll
    for (int mi = 0; mi < 2; ++mi)
#pragma unroll
      for (int ni = 0; ni < 2; ++ni)
        acc[mi][ni] = __builtin_amdgcn_wmma_f32_16x16x32_bf16(
            false, fa[mi].v, false, fb[ni].v, (short)0, acc[mi][ni],
            false, false);
  };

  const int nkt = (K + TK - 1) / TK;
  stage(0, 0);
  asm volatile("s_wait_asynccnt 0x0" ::: "memory");
  __syncthreads();
  for (int kt = 0; kt < nkt; ++kt) {
    if (kt + 1 < nkt) stage((kt + 1) & 1, (kt + 1) * TK); // overlap w/ compute
    compute(kt & 1);
    asm volatile("s_wait_asynccnt 0x0" ::: "memory");      // my copies landed
    __syncthreads();
  }

  // C/D layout: lanes 0-15: M=r, lanes 16-31: M=r+8; N = lane%16
#pragma unroll
  for (int mi = 0; mi < 2; ++mi) {
#pragma unroll
    for (int ni = 0; ni < 2; ++ni) {
      int cn = bN + (wNi << 5) + (ni << 4) + l16;
#pragma unroll
      for (int r = 0; r < 8; ++r) {
        int cm = bM + (wMi << 5) + (mi << 4) + r + (half << 3);
        if (cm < M && cn < N) {
          float v = acc[mi][ni][r];
          if (bias) v += bias[cn];
          if (act == 1) v = fmaxf(v, 0.0f);
          else if (act == 2) v = tanhf(v);
          C[(size_t)cm * N + cn] = v;
        }
      }
    }
  }
}

// =====================================================================
// Conversion / prep kernels
// =====================================================================
__global__ void cvt_f32_bf16(const float* __restrict__ x, bf16* __restrict__ y,
                             int n2) {  // n/2 pairs
  int i = blockIdx.x * blockDim.x + threadIdx.x;
  if (i >= n2) return;
  float2 v = ((const float2*)x)[i];
  ((unsigned*)y)[i] = pack_bf16(v.x, v.y);
}

// Bt[n*K+k] = bf16(B[k*N+n]) : transpose + convert (weights; small)
__global__ void wt_cvt_t(const float* __restrict__ B, bf16* __restrict__ Bt,
                         int K, int N) {
  int i = blockIdx.x * blockDim.x + threadIdx.x;
  if (i >= K * N) return;
  int n = i / K, k = i - n * K;
  Bt[(size_t)n * K + k] = (bf16)B[(size_t)k * N + n];
}

// y32 = relu(x); y16 = bf16(y32)   (dual output, vectorized by pairs)
__global__ void relu_dual(const float* __restrict__ x, float* __restrict__ y32,
                          bf16* __restrict__ y16, int n2) {
  int i = blockIdx.x * blockDim.x + threadIdx.x;
  if (i >= n2) return;
  float2 v = ((const float2*)x)[i];
  v.x = fmaxf(v.x, 0.f); v.y = fmaxf(v.y, 0.f);
  ((float2*)y32)[i] = v;
  ((unsigned*)y16)[i] = pack_bf16(v.x, v.y);
}

__global__ void fill_u32(unsigned* __restrict__ p, unsigned v, int n) {
  int i = blockIdx.x * blockDim.x + threadIdx.x;
  if (i < n) p[i] = v;
}

// wdv[k] = sum_j Wd[k,j] * a_d[j]   (fused zd @ a_d = X @ wdv)
__global__ void fuse_wd_ad(const float* __restrict__ Wd, const float* __restrict__ ad,
                           float* __restrict__ wdv, int fin, int fout) {
  int k = blockIdx.x * blockDim.x + threadIdx.x;
  if (k >= fin) return;
  float a = 0.f;
  for (int j = 0; j < fout; ++j) a += Wd[(size_t)k * fout + j] * ad[j];
  wdv[k] = a;
}

// out[row] = X[row,:] . w   (one wave per row; coalesced lane-strided loads)
__global__ void rowdot(const float* __restrict__ X, const float* __restrict__ w,
                       float* __restrict__ out, int n, int d) {
  int gt = blockIdx.x * blockDim.x + threadIdx.x;
  int row = gt >> 5, lane = gt & 31;
  if (row >= n) return;
  float a = 0.f;
  for (int k = lane; k < d; k += 32) a += X[(size_t)row * d + k] * w[k];
  for (int off = 16; off; off >>= 1) a += __shfl_down(a, off, 32);
  if (lane == 0) out[row] = a;
}

// order-preserving float<->uint for atomic max
__device__ __forceinline__ unsigned fenc(float f) {
  unsigned u = __float_as_uint(f);
  return (u >> 31) ? ~u : (u | 0x80000000u);
}
__device__ __forceinline__ float fdec(unsigned u) {
  return (u >> 31) ? __uint_as_float(u & 0x7fffffffu) : __uint_as_float(~u);
}

__global__ void edge_logit_max(int E, const float* __restrict__ ef, int ed,
                               const float* __restrict__ ae,
                               const int* __restrict__ src, const int* __restrict__ dst,
                               const float* __restrict__ s, const float* __restrict__ d,
                               float* __restrict__ logit, unsigned* __restrict__ menc) {
  int e = blockIdx.x * blockDim.x + threadIdx.x;
  if (e >= E) return;
  float a = s[src[e]] + d[dst[e]];
  for (int i = 0; i < ed; ++i) a += ef[(size_t)e * ed + i] * ae[i];
  a = (a > 0.f) ? a : LEAKY_SLOPE * a;   // leaky_relu
  logit[e] = a;
  atomicMax(&menc[dst[e]], fenc(a));
}

__global__ void edge_exp_sum(int E, const int* __restrict__ dst,
                             float* __restrict__ logit, const unsigned* __restrict__ menc,
                             float* __restrict__ ssum) {
  int e = blockIdx.x * blockDim.x + threadIdx.x;
  if (e >= E) return;
  int j = dst[e];
  float ex = __expf(logit[e] - fdec(menc[j]));
  logit[e] = ex;                          // reuse buffer for ex
  atomicAdd(&ssum[j], ex);
}

// out[dst] += (ex/(sum[dst]+1e-9)) * zs[src]   (2 floats per thread)
__global__ void edge_scatter(int E, int fout, const float* __restrict__ ex,
                             const float* __restrict__ ssum,
                             const int* __restrict__ src, const int* __restrict__ dst,
                             const float* __restrict__ zs, float* __restrict__ out) {
  int t = blockIdx.x * blockDim.x + threadIdx.x;
  int hw = fout >> 1;
  if (t >= E * hw) return;
  int e = t / hw;
  int f = (t - e * hw) << 1;
  int sj = src[e], dj = dst[e];
  float alpha = ex[e] / (ssum[dj] + 1e-9f);
  float2 z = *(const float2*)&zs[(size_t)sj * fout + f];
  atomicAdd(&out[(size_t)dj * fout + f + 0], alpha * z.x);
  atomicAdd(&out[(size_t)dj * fout + f + 1], alpha * z.y);
}

// e_out = relu(e_in @ We + be), d <= 5 (register-buffered -> in-place safe)
__global__ void edge_update(int E, int d, const float* __restrict__ We,
                            const float* __restrict__ be,
                            const float* __restrict__ ein, float* __restrict__ eout) {
  int e = blockIdx.x * blockDim.x + threadIdx.x;
  if (e >= E) return;
  float in[5], out[5];
  for (int i = 0; i < d; ++i) in[i] = ein[(size_t)e * d + i];
  for (int j = 0; j < d; ++j) {
    float a = be[j];
    for (int i = 0; i < d; ++i) a += in[i] * We[i * d + j];
    out[j] = fmaxf(a, 0.f);
  }
  for (int j = 0; j < d; ++j) eout[(size_t)e * d + j] = out[j];
}

// =====================================================================
// Host orchestration
// =====================================================================
struct Rel { const float *Ws, *Wd, *a_s, *a_d, *a_e, *We, *be; };

static inline int cdiv(long long a, long long b) { return (int)((a + b - 1) / b); }

extern "C" void kernel_launch(void* const* d_in, const int* in_sizes, int n_in,
                              void* d_out, int out_size, void* d_ws, size_t ws_size,
                              hipStream_t stream) {
  // ---- fixed inputs ----
  const float* XR0  = (const float*)d_in[0];
  const float* XF0  = (const float*)d_in[1];
  const float* EFF0 = (const float*)d_in[2];
  const float* ERR0 = (const float*)d_in[3];
  const float* ERF0 = (const float*)d_in[4];
  const int* ffs = (const int*)d_in[5]; const int* ffd = (const int*)d_in[6];
  const int* rrs = (const int*)d_in[7]; const int* rrd = (const int*)d_in[8];
  const int* rfs = (const int*)d_in[9]; const int* rfd = (const int*)d_in[10];

  // ---- param leaf indexing: sorted-key (jax tree_flatten) vs insertion order ----
  bool sortedOrder = (n_in > 12) ? (in_sizes[12] == 9) : true;
  auto P = [&](int i) -> const float* { return (const float*)d_in[i]; };
  auto getRel = [&](int layer, int relS /*0 ff,1 rf,2 rr*/) -> Rel {
    Rel r;
    if (sortedOrder) {
      int b = 11 + layer * 21 + relS * 7;
      r.Wd = P(b+0); r.We = P(b+1); r.Ws = P(b+2);
      r.a_d = P(b+3); r.a_e = P(b+4); r.a_s = P(b+5); r.be = P(b+6);
    } else {
      int relI = (relS == 0) ? 0 : (relS == 2 ? 1 : 2); // ff, rr, rf
      int b = 11 + layer * 21 + relI * 7;
      r.Ws = P(b+0); r.Wd = P(b+1); r.a_s = P(b+2);
      r.a_d = P(b+3); r.a_e = P(b+4); r.We = P(b+5); r.be = P(b+6);
    }
    return r;
  };
  const float *WmW, *WmB, *WvW, *WvB;
  {
    int tb = 11 + 63;
    if (sortedOrder) { WvW = P(tb); WvB = P(tb+1); WmW = P(tb+2); WmB = P(tb+3); }
    else             { WmW = P(tb); WmB = P(tb+1); WvW = P(tb+2); WvB = P(tb+3); }
  }

  // ---- workspace carve-up ----
  char* wb = (char*)d_ws; size_t off = 0;
  auto allocB = [&](size_t nbytes) -> void* {
    void* p = (void*)(wb + off);
    off += ((nbytes + 255) & ~(size_t)255);
    return p;
  };
  auto allocF = [&](size_t nf) -> float* { return (float*)allocB(nf * 4); };
  auto allocH = [&](size_t nh) -> bf16*  { return (bf16*)allocB(nh * 2); };

  float* xf_a = allocF((size_t)N_FURN * 256);
  float* xf_b = allocF((size_t)N_FURN * 256);
  float* m_f  = allocF((size_t)N_FURN * 256);
  float* zs_f = allocF((size_t)N_FURN * 256);
  float* zs_r = allocF((size_t)N_ROOM * 256);
  float* xr_a = allocF((size_t)N_ROOM * ROOM_D);
  float* xr_b = allocF((size_t)N_ROOM * ROOM_D);
  float* m_r  = allocF((size_t)N_ROOM * ROOM_D);
  float* wdv  = allocF(1040);
  float* sS   = allocF(N_FURN);
  float* dS   = allocF(N_FURN);
  unsigned* menc = (unsigned*)allocF(N_FURN);
  float* ssum = allocF(N_FURN);
  float* elog = allocF(NE_RF);
  float* eff  = allocF((size_t)NE_FF * 3);
  float* err  = allocF((size_t)NE_RR * 4);
  float* erf  = allocF((size_t)NE_RF * 5);
  // bf16 A operands + transposed weights
  bf16* xf0_16  = allocH((size_t)N_FURN * 1040);
  bf16* xf_a16  = allocH((size_t)N_FURN * 256);
  bf16* xf_b16  = allocH((size_t)N_FURN * 256);
  bf16* xr0_16  = allocH((size_t)N_ROOM * ROOM_D);
  bf16* xr_a16  = allocH((size_t)N_ROOM * ROOM_D);
  bf16* xr_b16  = allocH((size_t)N_ROOM * ROOM_D);
  bf16* btmp    = allocH((size_t)1040 * 256);

  // ---- launch helpers ----
  auto gemmT = [&](const bf16* A16, const float* W, float* C, int M, int N, int K,
                   const float* bias, int act) {
    wt_cvt_t<<<cdiv((long long)K * N, 256), 256, 0, stream>>>(W, btmp, K, N);
    dim3 g(cdiv(M, TM), cdiv(N, TN));
    gemm_bf16_wmma<<<g, 256, 0, stream>>>(A16, btmp, C, M, N, K, bias, act);
  };
  auto fill0 = [&](void* p, long long n) {
    fill_u32<<<cdiv(n, 256), 256, 0, stream>>>((unsigned*)p, 0u, (int)n);
  };
  auto relation = [&](const bf16* Xs16, int Ns,
                      const float* Xd32, int Nd, int finD,
                      const float* efeat, int E, int ed,
                      const int* src, const int* dst,
                      const Rel& r, int finS, int fout,
                      float* zsbuf, float* outAccum) {
    fuse_wd_ad<<<cdiv(finD, 256), 256, 0, stream>>>(r.Wd, r.a_d, wdv, finD, fout);
    gemmT(Xs16, r.Ws, zsbuf, Ns, fout, finS, nullptr, 0);
    rowdot<<<cdiv((long long)Ns * 32, 256), 256, 0, stream>>>(zsbuf, r.a_s, sS, Ns, fout);
    rowdot<<<cdiv((long long)Nd * 32, 256), 256, 0, stream>>>(Xd32, wdv, dS, Nd, finD);
    fill0(menc, Nd); fill0(ssum, Nd);
    edge_logit_max<<<cdiv(E, 256), 256, 0, stream>>>(E, efeat, ed, r.a_e, src, dst, sS, dS, elog, menc);
    edge_exp_sum<<<cdiv(E, 256), 256, 0, stream>>>(E, dst, elog, menc, ssum);
    long long st = (long long)E * (fout >> 1);
    edge_scatter<<<cdiv(st, 256), 256, 0, stream>>>(E, fout, elog, ssum, src, dst, zsbuf, outAccum);
  };

  // bf16 copies of layer-0 node features
  cvt_f32_bf16<<<cdiv((long long)N_FURN * 1040 / 2, 256), 256, 0, stream>>>(
      XF0, xf0_16, N_FURN * 1040 / 2);
  cvt_f32_bf16<<<cdiv((long long)N_ROOM * ROOM_D / 2, 256), 256, 0, stream>>>(
      XR0, xr0_16, N_ROOM * ROOM_D / 2);

  const int fins[3]  = {1040, 256, 128};
  const int fouts[3] = {256, 128, 64};
  const float* XF32 = XF0;  const bf16* XF16 = xf0_16;
  const float* XR32 = XR0;  const bf16* XR16 = xr0_16;
  const float* cEFF = EFF0; const float* cERR = ERR0; const float* cERF = ERF0;
  float* xfN32[3] = {xf_a, xf_b, xf_a};
  bf16*  xfN16[3] = {xf_a16, xf_b16, xf_a16};
  float* xrN32[2] = {xr_a, xr_b};
  bf16*  xrN16[2] = {xr_a16, xr_b16};

  for (int l = 0; l < 3; ++l) {
    const int fin = fins[l], fout = fouts[l];
    Rel rff = getRel(l, 0), rrf = getRel(l, 1), rrr = getRel(l, 2);

    // rr relation (room->room); layer-2 result is dead -> skip
    if (l < 2) {
      fill0(m_r, (long long)N_ROOM * ROOM_D);
      relation(XR16, N_ROOM, XR32, N_ROOM, ROOM_D, cERR, NE_RR, 4, rrs, rrd,
               rrr, ROOM_D, ROOM_D, zs_r, m_r);
      relu_dual<<<cdiv((long long)N_ROOM * ROOM_D / 2, 256), 256, 0, stream>>>(
          m_r, xrN32[l], xrN16[l], N_ROOM * ROOM_D / 2);
    }

    // ff + rf relations accumulate into m_f (furniture messages)
    fill0(m_f, (long long)N_FURN * fout);
    relation(XF16, N_FURN, XF32, N_FURN, fin, cEFF, NE_FF, 3, ffs, ffd,
             rff, fin, fout, zs_f, m_f);
    relation(XR16, N_ROOM, XF32, N_FURN, fin, cERF, NE_RF, 5, rfs, rfd,
             rrf, ROOM_D, fout, zs_r, m_f);
    relu_dual<<<cdiv((long long)N_FURN * fout / 2, 256), 256, 0, stream>>>(
        m_f, xfN32[l], xfN16[l], N_FURN * fout / 2);

    // edge feature updates (only where consumed downstream)
    if (l < 2) {
      edge_update<<<cdiv(NE_FF, 256), 256, 0, stream>>>(NE_FF, 3, rff.We, rff.be, cEFF, eff);
      edge_update<<<cdiv(NE_RF, 256), 256, 0, stream>>>(NE_RF, 5, rrf.We, rrf.be, cERF, erf);
      if (l < 1)
        edge_update<<<cdiv(NE_RR, 256), 256, 0, stream>>>(NE_RR, 4, rrr.We, rrr.be, cERR, err);
      cEFF = eff; cERF = erf; cERR = err;
    }
    XF32 = xfN32[l]; XF16 = xfN16[l];
    if (l < 2) { XR32 = xrN32[l]; XR16 = xrN16[l]; }
  }

  // final heads: tanh(XF @ W + b), XF = [50000, 64]
  float* mu = (float*)d_out;
  float* lv = (float*)d_out + (size_t)N_FURN * 64;
  gemmT(XF16, WmW, mu, N_FURN, 64, 64, WmB, 2);
  gemmT(XF16, WvW, lv, N_FURN, 64, 64, WvB, 2);
}
